// EdgeAttentionEmbedding_71339406786694
// MI455X (gfx1250) — matrix operand order repaired and checked by reading, not compiled, not run on hardware
//
#include <hip/hip_runtime.h>
#include <hip/hip_bf16.h>

typedef __attribute__((ext_vector_type(16))) _Float16 v16h;
typedef __attribute__((ext_vector_type(8)))  float    v8f;

#define DN_IN 128
#define DE_IN 64
#define KTOT  192   // DN_IN + DE_IN
#define DOUT  128
#define NFRAG (6*8*32*16)   // kc * ntile * lane * elems  = 24576 halves = 48KB

// ---------------------------------------------------------------------------
// Kernel 1: fold the per-head linear chain into one fused weight + bias.
//   Wc[k][n] = 0.5 * sum_h ( k<128 ? (W1h @ W3h_top)[k][n] : W3h[k][n] )
//   bc[n]    = 0.5 * sum_h ( 2*b1h @ W3h_top + b3h )[n]
// Wc is written directly in WMMA B-fragment order (f16) so the main kernel's
// LDS reads are trivially one 32B vector per lane.
// ---------------------------------------------------------------------------
__global__ __launch_bounds__(128)
void eae_precompute(const float* __restrict__ W1, const float* __restrict__ b1,
                    const float* __restrict__ W3, const float* __restrict__ b3,
                    _Float16* __restrict__ Bfrag, float* __restrict__ bc) {
    const int n = threadIdx.x;      // 0..127 (output column)
    const int k = blockIdx.x;       // 0..191 weights, 192 -> bias
    if (k < KTOT) {
        float acc = 0.0f;
        if (k < DN_IN) {
            #pragma unroll 1
            for (int h = 0; h < 2; ++h) {
                const float* w1r = W1 + ((size_t)h * DN_IN + k) * DN_IN;  // W1[h][k][:]
                const float* w3c = W3 + (size_t)h * KTOT * DOUT + n;       // W3[h][:][n]
                for (int j = 0; j < DN_IN; ++j)
                    acc += w1r[j] * w3c[(size_t)j * DOUT];
            }
        } else {
            acc = W3[(size_t)k * DOUT + n] + W3[(size_t)(KTOT + k) * DOUT + n];
        }
        acc *= 0.5f;
        // B-fragment swizzle: chunk kc covers K = 32*kc .. 32*kc+31
        const int kc   = k >> 5;
        const int krel = k & 31;
        const int half = (krel >> 3) & 1;   // lanes 0-15 hold K 0..7,16..23, lanes 16-31 hold 8..15,24..31
        const int i    = (krel & 7) | ((krel >> 4) << 3);
        const int nt   = n >> 4;
        const int lane = half * 16 + (n & 15);
        Bfrag[(((kc * 8 + nt) * 32) + lane) * 16 + i] = (_Float16)acc;
    } else {
        float acc = b3[n] + b3[DOUT + n];
        #pragma unroll 1
        for (int h = 0; h < 2; ++h) {
            const float* b1h = b1 + h * DN_IN;
            const float* w3c = W3 + (size_t)h * KTOT * DOUT + n;
            for (int j = 0; j < DN_IN; ++j)
                acc += 2.0f * b1h[j] * w3c[(size_t)j * DOUT];
        }
        bc[n] = 0.5f * acc;
    }
}

// ---------------------------------------------------------------------------
// Kernel 2: per-edge fused GEMM + row softmax.
// Each wave owns a 16-edge tile. k-chunk-outer / n-tile-inner ordering keeps
// exactly one A fragment (8 VGPRs) and the 8 f32 accumulators (64 VGPRs)
// live; B fragments are re-read from LDS each tile (optimization barrier
// prevents the compiler from hoisting 384 VGPRs of invariant weights).
// ---------------------------------------------------------------------------
__global__ __launch_bounds__(256, 4)
void eae_main(const float* __restrict__ nf, const float* __restrict__ efeat,
              const int* __restrict__ eidx, const _Float16* __restrict__ Bfrag,
              const float* __restrict__ bc, float* __restrict__ out, int E) {
    __shared__ __align__(32) _Float16 smB[NFRAG];   // 48 KB fused weight, fragment order

    const int tid = threadIdx.x;
    // cooperative 48KB weight copy (b128 loads/stores)
    {
        const float4* s = (const float4*)Bfrag;
        float4* d = (float4*)smB;
        #pragma unroll 4
        for (int i = tid; i < NFRAG * 2 / 16; i += 256) d[i] = s[i];
    }
    __syncthreads();

    const int lane = tid & 31;
    const int wid  = tid >> 5;
    const int half = lane >> 4;   // which 16-lane half
    const int ml   = lane & 15;

    // bias for this lane's C column (N = ml), per n-tile — invariant across tiles
    float biasv[8];
    #pragma unroll
    for (int nt = 0; nt < 8; ++nt) biasv[nt] = bc[nt * 16 + ml];

    const int ntiles = (E + 15) >> 4;
    const int stride = gridDim.x * 8;

    for (int tile = blockIdx.x * 8 + wid; tile < ntiles; tile += stride) {
        // optimization barrier: forbid hoisting the invariant LDS weight
        // fragments (384 VGPRs worth) out of the tile loop
        asm volatile("" ::: "memory");

        const int e0 = tile << 4;
        int ea  = e0 + ml;
        int eaL = ea < E ? ea : E - 1;
        const int u = eidx[eaL];
        const int v = eidx[E + eaL];
        const float cfac = (ea < E) ? ((u == v) ? 1.0f : 0.5f) : 0.0f;
        const float4* hu = (const float4*)(nf + (size_t)u * DN_IN);
        const float4* hv = (const float4*)(nf + (size_t)v * DN_IN);
        const float4* ef = (const float4*)(efeat + (size_t)eaL * DE_IN);

        // 8 accumulators = the 16x128 logit tile (initialized with bias)
        v8f acc[8];
        #pragma unroll
        for (int nt = 0; nt < 8; ++nt)
            #pragma unroll
            for (int r = 0; r < 8; ++r) acc[nt][r] = biasv[nt];

        #pragma unroll
        for (int kc = 0; kc < 6; ++kc) {
            // ---- build the single A fragment for this k-chunk
            // A layout: lane holds row M=ml; elems 0-7 <-> K=8*half+0..7,
            // elems 8-15 <-> K=16+8*half+0..7 (within the 32-wide chunk)
            v16h af;
            if (kc < 4) {                       // node part: m = cfac*(h_u+h_v)
                const int b4 = kc * 8 + half * 2;
                #pragma unroll
                for (int q = 0; q < 2; ++q) {
                    float4 xu = hu[b4 + q],     xv = hv[b4 + q];
                    float4 yu = hu[b4 + 4 + q], yv = hv[b4 + 4 + q];
                    af[4*q+0] = (_Float16)(cfac * (xu.x + xv.x));
                    af[4*q+1] = (_Float16)(cfac * (xu.y + xv.y));
                    af[4*q+2] = (_Float16)(cfac * (xu.z + xv.z));
                    af[4*q+3] = (_Float16)(cfac * (xu.w + xv.w));
                    af[8+4*q+0] = (_Float16)(cfac * (yu.x + yv.x));
                    af[8+4*q+1] = (_Float16)(cfac * (yu.y + yv.y));
                    af[8+4*q+2] = (_Float16)(cfac * (yu.z + yv.z));
                    af[8+4*q+3] = (_Float16)(cfac * (yu.w + yv.w));
                }
            } else {                            // edge-feature part
                const int b4 = (kc - 4) * 8 + half * 2;
                #pragma unroll
                for (int q = 0; q < 2; ++q) {
                    float4 x = ef[b4 + q], y = ef[b4 + 4 + q];
                    af[4*q+0] = (_Float16)x.x;  af[4*q+1] = (_Float16)x.y;
                    af[4*q+2] = (_Float16)x.z;  af[4*q+3] = (_Float16)x.w;
                    af[8+4*q+0] = (_Float16)y.x; af[8+4*q+1] = (_Float16)y.y;
                    af[8+4*q+2] = (_Float16)y.z; af[8+4*q+3] = (_Float16)y.w;
                }
            }
            // ---- 8 n-tiles against this k-chunk
            #pragma unroll
            for (int nt = 0; nt < 8; ++nt) {
                const v16h bf = *(const v16h*)(smB + (((kc * 8 + nt) * 32) + lane) * 16);
                acc[nt] = __builtin_amdgcn_wmma_f32_16x16x32_f16(
                            false, af, false, bf, (short)0, acc[nt], false, false);
            }
        }

        // ---- fused row softmax: row = e0 + r + 8*half, N spread over (nt, ml)
        #pragma unroll
        for (int r = 0; r < 8; ++r) {
            float mx = acc[0][r];
            #pragma unroll
            for (int nt = 1; nt < 8; ++nt) mx = fmaxf(mx, acc[nt][r]);
            #pragma unroll
            for (int m = 1; m < 16; m <<= 1) mx = fmaxf(mx, __shfl_xor(mx, m, 32));
            float ssum = 0.0f;
            #pragma unroll
            for (int nt = 0; nt < 8; ++nt) {
                float ex = __expf(acc[nt][r] - mx);
                acc[nt][r] = ex;
                ssum += ex;
            }
            #pragma unroll
            for (int m = 1; m < 16; m <<= 1) ssum += __shfl_xor(ssum, m, 32);
            const float inv = 1.0f / ssum;
            const int row = e0 + r + 8 * half;
            if (row < E) {
                float* op = out + (size_t)row * DOUT + ml;
                #pragma unroll
                for (int nt = 0; nt < 8; ++nt) op[nt * 16] = acc[nt][r] * inv;
            }
        }
    }
}

extern "C" void kernel_launch(void* const* d_in, const int* in_sizes, int n_in,
                              void* d_out, int out_size, void* d_ws, size_t ws_size,
                              hipStream_t stream) {
    const float* nf    = (const float*)d_in[0];
    const float* efeat = (const float*)d_in[1];
    const int*   eidx  = (const int*)  d_in[2];
    const float* W1    = (const float*)d_in[3];
    const float* b1    = (const float*)d_in[4];
    // d_in[5]=W2, d_in[6]=b2, d_in[9]=Wa, d_in[10]=ba are mathematically dead
    const float* W3    = (const float*)d_in[7];
    const float* b3    = (const float*)d_in[8];
    const int E = in_sizes[2] / 2;

    _Float16* Bfrag = (_Float16*)d_ws;
    float*    bc    = (float*)((char*)d_ws + NFRAG * sizeof(_Float16));

    eae_precompute<<<KTOT + 1, 128, 0, stream>>>(W1, b1, W3, b3, Bfrag, bc);

    const int ntiles = (E + 15) / 16;
    int blocks = (ntiles + 7) / 8;
    if (blocks > 1024) blocks = 1024;
    eae_main<<<blocks, 256, 0, stream>>>(nf, efeat, eidx, Bfrag, bc, (float*)d_out, E);
}
